// TransformerEncoderLayer_75419625717967
// MI455X (gfx1250) — matrix operand order repaired and checked by reading, not compiled, not run on hardware
//
#include <hip/hip_runtime.h>
#include <math.h>

// ---------------------------------------------------------------------------
// Types / constants
// ---------------------------------------------------------------------------
typedef __bf16 bf16_t;
typedef __attribute__((ext_vector_type(16))) __bf16 v16bf;
typedef __attribute__((ext_vector_type(8)))  float  v8f;
typedef __attribute__((ext_vector_type(4)))  unsigned int u32x4;
typedef __attribute__((ext_vector_type(8)))  int i32x8;
typedef __attribute__((ext_vector_type(4)))  int i32x4;

struct alignas(16) BF8 { bf16_t v[8]; };

#define USE_TDM 1    // Tensor Data Mover staging of A tiles (double-buffered)

#define TB   256     // threads per block (8 waves, wave32)
#define BM   128
#define BN   64
#define BK   32
#define LSTR (BK + 8)   // LDS row stride in bf16: 32 data + 8 pad = 80B rows

#define B_   8
#define L_   1024
#define D_   256
#define H_   8
#define HS_  32
#define FFN_ 1024
#define KW_  17
#define DP_  64
#define EPS_ 1e-5f

// ---------------------------------------------------------------------------
// fp32 -> bf16 convert
// ---------------------------------------------------------------------------
__global__ __launch_bounds__(256) void cvt_f32_bf16(const float* __restrict__ s,
                                                    bf16_t* __restrict__ d,
                                                    long long n) {
  long long i = (long long)blockIdx.x * 256 + threadIdx.x;
  if (i < n) d[i] = (bf16_t)s[i];
}

// ---------------------------------------------------------------------------
// Dynamic position bias MLP: table[p][h], p in [0, 2L-1)
// ---------------------------------------------------------------------------
__global__ __launch_bounds__(256) void dpb_kernel(
    const float* __restrict__ w0, const float* __restrict__ b0,
    const float* __restrict__ w1, const float* __restrict__ b1,
    const float* __restrict__ w2, const float* __restrict__ b2,
    float* __restrict__ table) {
  int p = blockIdx.x * 256 + threadIdx.x;
  if (p >= 2 * L_ - 1) return;
  float pos = (float)(p - (L_ - 1));
  float h0[DP_], h1[DP_];
  for (int j = 0; j < DP_; ++j) {
    float t = pos * w0[j] + b0[j];
    h0[j] = t * (1.f / (1.f + __expf(-t)));           // SiLU
  }
  for (int j = 0; j < DP_; ++j) {
    float s = b1[j];
    for (int i = 0; i < DP_; ++i) s += h0[i] * w1[i * DP_ + j];
    h1[j] = s * (1.f / (1.f + __expf(-s)));
  }
  for (int h = 0; h < H_; ++h) {
    float s = b2[h];
    for (int i = 0; i < DP_; ++i) s += h1[i] * w2[i * H_ + h];
    table[p * H_ + h] = s;
  }
}

// ---------------------------------------------------------------------------
// TDM: issue a tensor_load_to_lds for one A tile (BM rows x BK bf16).
// The LDS pad (pad_interval=16 DWORDs data, pad_amount=4 DWORDs pad) exactly
// reproduces the BK+8 LDS row stride used by the WMMA fragment loads.
// Descriptor per CDNA5 ISA.
// ---------------------------------------------------------------------------
__device__ __forceinline__ void tdm_load_a_tile(const bf16_t* gsrc,
                                                unsigned lds_off,
                                                int rows_rem, int cols_rem,
                                                int lda) {
  unsigned long long ga = (unsigned long long)(uintptr_t)gsrc;
  u32x4 g0;
  g0[0] = 1u;                                           // count=1 (valid D#)
  g0[1] = lds_off;                                      // lds_addr (bytes)
  g0[2] = (unsigned)(ga & 0xffffffffu);                 // global_addr[31:0]
  g0[3] = (unsigned)((ga >> 32) & 0x01ffffffu)          // global_addr[56:32]
        | (2u << 30);                                   // type = 2 ("image")
  i32x8 g1;
  g1[0] = (1 << 16)        // data_size = 2 bytes
        | (1 << 20)        // pad_enable
        | (3 << 22)        // pad_interval: 16 DWORDs (= one 32-elem bf16 row)
        | (3 << 25);       // pad_amount: 4 DWORDs (= 8 bf16) -> LSTR stride
  g1[1] = (cols_rem & 0xffff) << 16;                    // tensor_dim0[15:0]
  g1[2] = ((unsigned)cols_rem >> 16)                    // tensor_dim0[31:16]
        | ((rows_rem & 0xffff) << 16);                  // tensor_dim1[15:0]
  g1[3] = ((unsigned)rows_rem >> 16)                    // tensor_dim1[31:16]
        | (BK << 16);                                   // tile_dim0 = 32
  g1[4] = BM;                                           // tile_dim1 = 128
  g1[5] = lda;                                          // dim0 stride (elems)
  g1[6] = 0;
  g1[7] = 0;
  i32x4 gz = {0, 0, 0, 0};
  asm volatile("tensor_load_to_lds %0, %1, %2, %3"
               :: "s"(g0), "s"(g1), "s"(gz), "s"(gz)
               : "memory");
}

// ---------------------------------------------------------------------------
// Stage B tile transposed into Bs[n][k] (bf16).
// ---------------------------------------------------------------------------
__device__ __forceinline__ void stage_B_tile(const bf16_t* __restrict__ Bm,
                                             bf16_t (*Bs)[LSTR], int k0, int N,
                                             int ldb, int tileN, int transB,
                                             bool fullN, int tid) {
  if (transB) {  // B stored NxK row-major
    if (fullN) {
      for (int c = tid; c < (BN * BK / 8); c += TB) {
        int n = c >> 2, kc = (c & 3) * 8;
        *(BF8*)&Bs[n][kc] =
            *(const BF8*)(Bm + (long long)(tileN + n) * ldb + k0 + kc);
      }
    } else {
      for (int c = tid; c < (BN * BK / 8); c += TB) {
        int n = c >> 2, kc = (c & 3) * 8;
        int gn = tileN + n;
        BF8 t;
        if (gn < N) t = *(const BF8*)(Bm + (long long)gn * ldb + k0 + kc);
        else        for (int j = 0; j < 8; ++j) t.v[j] = (bf16_t)0.f;
        *(BF8*)&Bs[n][kc] = t;
      }
    }
  } else {       // B stored KxN row-major: transpose on the way into LDS
    for (int c = tid; c < (BN * BK / 8); c += TB) {
      int kk = c >> 3, n0 = (c & 7) * 8;
      if (fullN) {
        BF8 t = *(const BF8*)(Bm + (long long)(k0 + kk) * ldb + tileN + n0);
        for (int j = 0; j < 8; ++j) Bs[n0 + j][kk] = t.v[j];
      } else {
        for (int j = 0; j < 8; ++j) {
          int gn = tileN + n0 + j;
          Bs[n0 + j][kk] =
              (gn < N) ? Bm[(long long)(k0 + kk) * ldb + gn] : (bf16_t)0.f;
        }
      }
    }
  }
}

// ---------------------------------------------------------------------------
// Fallback VALU staging of an A tile (partial M tiles only).
// ---------------------------------------------------------------------------
__device__ __forceinline__ void stage_A_tile(const bf16_t* __restrict__ A,
                                             bf16_t (*As)[LSTR], int k0, int M,
                                             int lda, int tileM, int tid) {
  for (int c = tid; c < (BM * BK / 8); c += TB) {
    int r = c >> 2, kc = (c & 3) * 8;
    int gr = tileM + r;
    BF8 t;
    if (gr < M) t = *(const BF8*)(A + (long long)gr * lda + k0 + kc);
    else        for (int j = 0; j < 8; ++j) t.v[j] = (bf16_t)0.f;
    *(BF8*)&As[r][kc] = t;
  }
}

// ---------------------------------------------------------------------------
// One K-step of 2x2 WMMA tiles from LDS fragments.
// A 16x32 bf16 fragment: lane-half selects K 0..7 / 8..15 within each 16-chunk.
// B fragment from Bs[n][k]: same K order as A (lane-half selects K 0..7 / 8..15 within each 16-chunk).
// ---------------------------------------------------------------------------
__device__ __forceinline__ void mma_step(const bf16_t (*As)[LSTR],
                                         const bf16_t (*Bs)[LSTR],
                                         v8f acc[2][2], int wm, int wn,
                                         int rsel, int kbA, int khB) {
  for (int i = 0; i < 2; ++i) {
    v16bf a;
    const bf16_t* ap = &As[wm + i * 16 + rsel][0];
    for (int j = 0; j < 8; ++j) { a[j] = ap[kbA + j]; a[8 + j] = ap[16 + kbA + j]; }
    for (int j = 0; j < 2; ++j) {
      v16bf b;
      const bf16_t* bp = &Bs[wn + j * 16 + rsel][0];
      for (int q = 0; q < 16; ++q) b[q] = bp[kbA + (q & 7) + ((q >> 3) << 4)];
      acc[i][j] = __builtin_amdgcn_wmma_f32_16x16x32_bf16(
          false, a, false, b, (short)0, acc[i][j], false, false);
    }
  }
}

// ---------------------------------------------------------------------------
// Generic batched bf16 WMMA GEMM:  C = act(alpha * A@B + bias)
//  A: MxK row-major (lda).  B: KxN row-major (ldb) or NxK if transB.
//  Output fp32 (Cf) or bf16 (Cb), ldc.  Batch z offsets (z/zdiv, z%zdiv).
//  act: 0 = none, 1 = exact GELU.
//  A tiles staged by the Tensor Data Mover, double-buffered against WMMA.
// ---------------------------------------------------------------------------
__global__ __launch_bounds__(TB) void gemm_bf16_wmma(
    const bf16_t* __restrict__ A, const bf16_t* __restrict__ Bm,
    float* __restrict__ Cf, bf16_t* __restrict__ Cb,
    const float* __restrict__ bias,
    int M, int N, int K, int lda, int ldb, int ldc,
    int transB, int act, float alpha, int zdiv,
    long long oAo, long long oAi, long long oBo, long long oBi,
    long long oCo, long long oCi) {
  __shared__ bf16_t As[2][BM][LSTR];
  __shared__ bf16_t Bs[BN][LSTR];

  int z  = blockIdx.z;
  int zo = z / zdiv, zi = z % zdiv;
  A  += (long long)zo * oAo + (long long)zi * oAi;
  Bm += (long long)zo * oBo + (long long)zi * oBi;
  long long coff = (long long)zo * oCo + (long long)zi * oCi;

  int tileM = blockIdx.y * BM, tileN = blockIdx.x * BN;
  int tid = threadIdx.x, lane = tid & 31, wid = tid >> 5;
  int wm = (wid & 3) * 32;   // 4 wave rows
  int wn = (wid >> 2) * 32;  // 2 wave cols
  int rsel = lane & 15, hi = lane >> 4;
  int kbA = hi * 8;
  int khB = hi * 16;
  bool fullM = (tileM + BM <= M);           // uniform per block
  bool fullN = (tileN + BN <= N);           // uniform per block
  int  nk = (K + BK - 1) / BK;

  v8f acc[2][2] = {};

#if USE_TDM
  if (fullM) {
    const bf16_t* Abase = A + (long long)tileM * lda;
    if (wid == 0)
      tdm_load_a_tile(Abase, (unsigned)(uintptr_t)&As[0][0][0], M - tileM, K,
                      lda);
    int buf = 0;
    for (int kt = 0; kt < nk; ++kt) {
      int k0 = kt * BK;
      stage_B_tile(Bm, Bs, k0, N, ldb, tileN, transB, fullN, tid);
      if (kt + 1 < nk) {
        // prefetch next B K-slice while the TDM streams A
        __builtin_prefetch(
            Bm + (transB ? (long long)(tileN + (tid >> 2)) * ldb + k0 + BK
                         : (long long)(k0 + BK + (tid >> 3)) * ldb + tileN),
            0, 1);
      }
      if (wid == 0) {
        if (kt + 1 < nk) {
          tdm_load_a_tile(Abase + (k0 + BK),
                          (unsigned)(uintptr_t)&As[buf ^ 1][0][0], M - tileM,
                          K - (k0 + BK), lda);
          __builtin_amdgcn_s_wait_tensorcnt(1);  // tile kt done, kt+1 in flight
        } else {
          __builtin_amdgcn_s_wait_tensorcnt(0);
        }
      }
      __syncthreads();
      mma_step(As[buf], Bs, acc, wm, wn, rsel, kbA, khB);
      __syncthreads();
      buf ^= 1;
    }
  } else
#endif
  {
    for (int kt = 0; kt < nk; ++kt) {
      int k0 = kt * BK;
      stage_A_tile(A, As[0], k0, M, lda, tileM, tid);
      stage_B_tile(Bm, Bs, k0, N, ldb, tileN, transB, fullN, tid);
      __syncthreads();
      mma_step(As[0], Bs, acc, wm, wn, rsel, kbA, khB);
      __syncthreads();
    }
  }

  // ---- epilogue ----
  for (int i = 0; i < 2; ++i)
    for (int j = 0; j < 2; ++j) {
      int col = tileN + wn + j * 16 + rsel;
      if (col >= N) continue;
      float bv = bias ? bias[col] : 0.f;
      for (int v = 0; v < 8; ++v) {
        int row = tileM + wm + i * 16 + v + hi * 8;
        if (row >= M) continue;
        float val = acc[i][j][v] * alpha + bv;
        if (act == 1) val = 0.5f * val * (1.f + erff(val * 0.7071067811865476f));
        long long idx = coff + (long long)row * ldc + col;
        if (Cf) Cf[idx] = val;
        else    Cb[idx] = (bf16_t)val;
      }
    }
}

// ---------------------------------------------------------------------------
// Block reductions (LDS)
// ---------------------------------------------------------------------------
__device__ __forceinline__ float block_sum(float v, float* sb) {
  int t = threadIdx.x;
  sb[t] = v; __syncthreads();
  for (int s = 128; s > 0; s >>= 1) { if (t < s) sb[t] += sb[t + s]; __syncthreads(); }
  float r = sb[0]; __syncthreads();
  return r;
}
__device__ __forceinline__ float block_max(float v, float* sb) {
  int t = threadIdx.x;
  sb[t] = v; __syncthreads();
  for (int s = 128; s > 0; s >>= 1) { if (t < s) sb[t] = fmaxf(sb[t], sb[t + s]); __syncthreads(); }
  float r = sb[0]; __syncthreads();
  return r;
}

// ---------------------------------------------------------------------------
// Softmax over j with pos-bias gather + adj + mask.
// raw: [B,H,L,L] fp32 (already *1/sqrt(HS)).  Out: bf16 attn.
// ---------------------------------------------------------------------------
__global__ __launch_bounds__(256) void softmax_kernel(
    const float* __restrict__ raw, const float* __restrict__ adj,
    const float* __restrict__ table, const unsigned char* __restrict__ mask,
    bf16_t* __restrict__ attn) {
  __shared__ float sb[256];
  int i = blockIdx.x, z = blockIdx.y;
  int b = z >> 3, h = z & 7;
  const float* rp = raw + ((long long)z * L_ + i) * L_;
  const float* ap = adj + ((long long)b * L_ + i) * L_;
  const unsigned char* mk = mask + (long long)b * L_;

  float sv[4], mx = -3.402823466e38f;
  for (int q = 0; q < 4; ++q) {
    int j = threadIdx.x + q * 256;
    float s = rp[j] + table[(i - j + L_ - 1) * H_ + h] + ap[j];
    if (!mk[j]) s = -3.402823466e38f;
    sv[q] = s;
    mx = fmaxf(mx, s);
  }
  float m = block_max(mx, sb);
  float sum = 0.f;
  for (int q = 0; q < 4; ++q) { float e = __expf(sv[q] - m); sv[q] = e; sum += e; }
  float inv = 1.f / block_sum(sum, sb);
  bf16_t* op = attn + ((long long)z * L_ + i) * L_;
  for (int q = 0; q < 4; ++q) op[threadIdx.x + q * 256] = (bf16_t)(sv[q] * inv);
}

// ---------------------------------------------------------------------------
// h = LN(mask?(a):0 + b + biasvec).  Optionally second LN -> bf16 out, or
// plain bf16 copy of h.  One token per block, D_=256 threads.
// ---------------------------------------------------------------------------
__global__ __launch_bounds__(256) void resid_ln_kernel(
    const float* __restrict__ a, const float* __restrict__ b,
    const float* __restrict__ biasv, const unsigned char* __restrict__ maskp,
    const float* __restrict__ g1, const float* __restrict__ be1,
    const float* __restrict__ g2, const float* __restrict__ be2,
    float* __restrict__ hout, bf16_t* __restrict__ lnout) {
  __shared__ float s1[256];
  __shared__ float s2[256];
  long long t = blockIdx.x;
  int d = threadIdx.x;
  long long idx = t * D_ + d;

  float av = a[idx];
  if (maskp && !maskp[t]) av = 0.f;
  float v = av + (b ? b[idx] : 0.f) + (biasv ? biasv[d] : 0.f);

  int tid = threadIdx.x;
  s1[tid] = v; s2[tid] = v * v; __syncthreads();
  for (int s = 128; s > 0; s >>= 1) {
    if (tid < s) { s1[tid] += s1[tid + s]; s2[tid] += s2[tid + s]; }
    __syncthreads();
  }
  float mu = s1[0] * (1.f / D_);
  float var = s2[0] * (1.f / D_) - mu * mu;
  __syncthreads();
  float h = (v - mu) * rsqrtf(var + EPS_) * g1[d] + be1[d];
  if (hout) hout[idx] = h;

  if (g2) {
    s1[tid] = h; s2[tid] = h * h; __syncthreads();
    for (int s = 128; s > 0; s >>= 1) {
      if (tid < s) { s1[tid] += s1[tid + s]; s2[tid] += s2[tid + s]; }
      __syncthreads();
    }
    float mu2 = s1[0] * (1.f / D_);
    float var2 = s2[0] * (1.f / D_) - mu2 * mu2;
    float h2 = (h - mu2) * rsqrtf(var2 + EPS_) * g2[d] + be2[d];
    if (lnout) lnout[idx] = (bf16_t)h2;
  } else if (lnout) {
    lnout[idx] = (bf16_t)h;
  }
}

// ---------------------------------------------------------------------------
// GLU + depthwise conv (K=17, SAME, zero pad) + BN(eval) + Swish -> bf16.
// y: [token, 2D] fp32 from pw1 GEMM.  One (b,l) per block, threads = channel.
// ---------------------------------------------------------------------------
__global__ __launch_bounds__(256) void glu_conv_kernel(
    const float* __restrict__ y, const float* __restrict__ dw,
    const float* __restrict__ bng, const float* __restrict__ bnb,
    bf16_t* __restrict__ out) {
  int l = blockIdx.x, b = blockIdx.y, c = threadIdx.x;
  float acc = 0.f;
  for (int k = 0; k < KW_; ++k) {
    int lp = l + k - (KW_ / 2);
    if (lp < 0 || lp >= L_) continue;
    const float* yp = y + ((long long)(b * L_ + lp)) * (2 * D_) + c;
    float av = yp[0], gv = yp[D_];
    float glu = av * (1.f / (1.f + __expf(-gv)));
    acc += glu * dw[c * KW_ + k];
  }
  float v = acc * rsqrtf(1.f + EPS_) * bng[c] + bnb[c];
  v = v * (1.f / (1.f + __expf(-v)));   // Swish
  out[((long long)(b * L_ + l)) * D_ + c] = (bf16_t)v;
}

// ---------------------------------------------------------------------------
// Launcher
// ---------------------------------------------------------------------------
extern "C" void kernel_launch(void* const* d_in, const int* in_sizes, int n_in,
                              void* d_out, int out_size, void* d_ws, size_t ws_size,
                              hipStream_t stream) {
  (void)in_sizes; (void)n_in; (void)out_size; (void)ws_size;

  const float* x        = (const float*)d_in[0];
  const float* adj      = (const float*)d_in[1];
  const unsigned char* mask = (const unsigned char*)d_in[2];
  const float* w_qkv    = (const float*)d_in[3];
  const float* in_bias  = (const float*)d_in[4];
  const float* out_bias = (const float*)d_in[5];
  const float* dp_w0 = (const float*)d_in[6];
  const float* dp_b0 = (const float*)d_in[7];
  const float* dp_w1 = (const float*)d_in[8];
  const float* dp_b1 = (const float*)d_in[9];
  const float* dp_w2 = (const float*)d_in[10];
  const float* dp_b2 = (const float*)d_in[11];
  const float* ln1_g = (const float*)d_in[12];
  const float* ln1_b = (const float*)d_in[13];
  const float* ln2_g = (const float*)d_in[14];
  const float* ln2_b = (const float*)d_in[15];
  const float* ln3_g = (const float*)d_in[16];
  const float* ln3_b = (const float*)d_in[17];
  const float* ln4_g = (const float*)d_in[18];
  const float* ln4_b = (const float*)d_in[19];
  const float* f1_lg = (const float*)d_in[20];
  const float* f1_lb = (const float*)d_in[21];
  const float* f1_w1 = (const float*)d_in[22];
  const float* f1_b1 = (const float*)d_in[23];
  const float* f1_w2 = (const float*)d_in[24];
  const float* f1_b2 = (const float*)d_in[25];
  const float* f2_lg = (const float*)d_in[26];
  const float* f2_lb = (const float*)d_in[27];
  const float* f2_w1 = (const float*)d_in[28];
  const float* f2_b1 = (const float*)d_in[29];
  const float* f2_w2 = (const float*)d_in[30];
  const float* f2_b2 = (const float*)d_in[31];
  const float* pw1_w = (const float*)d_in[32];
  const float* pw1_b = (const float*)d_in[33];
  const float* dw_w  = (const float*)d_in[34];
  const float* bn_g  = (const float*)d_in[35];
  const float* bn_b  = (const float*)d_in[36];
  const float* pw2_w = (const float*)d_in[37];
  const float* pw2_b = (const float*)d_in[38];

  float* h4_out  = (float*)d_out;
  float* raw_out = (float*)d_out + (long long)B_ * L_ * D_;

  const long long TOK = (long long)B_ * L_;  // 8192 tokens

  size_t off = 0;
  auto alloc = [&](size_t bytes) -> void* {
    void* p = (char*)d_ws + off;
    off = (off + bytes + 255) & ~(size_t)255;
    return p;
  };

  bf16_t* xb     = (bf16_t*)alloc(TOK * D_ * 2);
  bf16_t* wqkvb  = (bf16_t*)alloc((size_t)D_ * 3 * D_ * 2);
  bf16_t* qkvb   = (bf16_t*)alloc(TOK * 3 * D_ * 2);
  float*  table  = (float*)alloc((size_t)(2 * L_ - 1) * H_ * 4);
  bf16_t* attnb  = (bf16_t*)alloc((size_t)B_ * H_ * L_ * L_ * 2);
  float*  attout = (float*)alloc(TOK * D_ * 4);
  float*  h1     = (float*)alloc(TOK * D_ * 4);
  bf16_t* h1ln   = (bf16_t*)alloc(TOK * D_ * 2);
  bf16_t* f1w1b  = (bf16_t*)alloc((size_t)D_ * FFN_ * 2);
  bf16_t* f1w2b  = (bf16_t*)alloc((size_t)FFN_ * D_ * 2);
  bf16_t* f2w1b  = (bf16_t*)alloc((size_t)D_ * FFN_ * 2);
  bf16_t* f2w2b  = (bf16_t*)alloc((size_t)FFN_ * D_ * 2);
  bf16_t* hidden = (bf16_t*)alloc(TOK * FFN_ * 2);
  float*  fout   = (float*)alloc(TOK * D_ * 4);
  float*  h2     = (float*)alloc(TOK * D_ * 4);
  bf16_t* h2b    = (bf16_t*)alloc(TOK * D_ * 2);
  bf16_t* pw1wb  = (bf16_t*)alloc((size_t)2 * D_ * D_ * 2);
  bf16_t* pw2wb  = (bf16_t*)alloc((size_t)D_ * D_ * 2);
  float*  pw1out = (float*)alloc(TOK * 2 * D_ * 4);
  bf16_t* convact= (bf16_t*)alloc(TOK * D_ * 2);
  float*  cv     = (float*)alloc(TOK * D_ * 4);
  float*  h3     = (float*)alloc(TOK * D_ * 4);
  bf16_t* h3ln   = (bf16_t*)alloc(TOK * D_ * 2);

  auto cvt = [&](const float* s, bf16_t* d, long long n) {
    cvt_f32_bf16<<<(unsigned)((n + 255) / 256), 256, 0, stream>>>(s, d, n);
  };
  auto gemm = [&](const bf16_t* A, const bf16_t* Bm, float* Cf, bf16_t* Cb,
                  const float* bias, int M, int N, int K, int lda, int ldb,
                  int ldc, int transB, int act, float alpha, int nz, int zdiv,
                  long long oAo, long long oAi, long long oBo, long long oBi,
                  long long oCo, long long oCi) {
    dim3 grid((N + BN - 1) / BN, (M + BM - 1) / BM, nz);
    gemm_bf16_wmma<<<grid, TB, 0, stream>>>(A, Bm, Cf, Cb, bias, M, N, K, lda,
                                            ldb, ldc, transB, act, alpha, zdiv,
                                            oAo, oAi, oBo, oBi, oCo, oCi);
  };

  // 1) conversions
  cvt(x, xb, TOK * D_);
  cvt(w_qkv, wqkvb, (long long)D_ * 3 * D_);
  cvt(f1_w1, f1w1b, (long long)D_ * FFN_);
  cvt(f1_w2, f1w2b, (long long)FFN_ * D_);
  cvt(f2_w1, f2w1b, (long long)D_ * FFN_);
  cvt(f2_w2, f2w2b, (long long)FFN_ * D_);
  cvt(pw1_w, pw1wb, (long long)2 * D_ * D_);   // [2D, D] used via transB
  cvt(pw2_w, pw2wb, (long long)D_ * D_);       // [D, D]  used via transB

  // 2) dynamic position bias table
  dpb_kernel<<<(2 * L_ - 1 + 255) / 256, 256, 0, stream>>>(dp_w0, dp_b0, dp_w1,
                                                           dp_b1, dp_w2, dp_b2,
                                                           table);

  // 3) QKV = x @ w_qkv + in_bias -> bf16 [token, 768]
  gemm(xb, wqkvb, nullptr, qkvb, in_bias, (int)TOK, 3 * D_, D_, D_, 3 * D_,
       3 * D_, 0, 0, 1.f, 1, 1, 0, 0, 0, 0, 0, 0);

  // 4) raw scores = (q @ k^T)/sqrt(HS) -> fp32 directly into d_out raw region.
  gemm(qkvb, qkvb + HS_, raw_out, nullptr, nullptr, L_, L_, HS_, 3 * D_,
       3 * D_, L_, 1, 0, 0.17677669529663687f, B_ * H_, H_,
       (long long)L_ * 3 * D_, 3 * HS_,
       (long long)L_ * 3 * D_, 3 * HS_,
       (long long)H_ * L_ * L_, (long long)L_ * L_);

  // 5) softmax(raw + posbias + adj, mask) -> bf16 attn
  softmax_kernel<<<dim3(L_, B_ * H_), 256, 0, stream>>>(raw_out, adj, table,
                                                        mask, attnb);

  // 6) out = attn @ v  -> fp32 [token, D] (heads interleaved via ldc/offsets)
  gemm(attnb, qkvb + 2 * HS_, attout, nullptr, nullptr, L_, HS_, L_, L_,
       3 * D_, D_, 0, 0, 1.f, B_ * H_, H_,
       (long long)H_ * L_ * L_, (long long)L_ * L_,
       (long long)L_ * 3 * D_, 3 * HS_,
       (long long)L_ * D_, HS_);

  // 7) h1 = LN(attout + out_bias + x); h1ln = LN(h1, f1_l) (bf16)
  resid_ln_kernel<<<(unsigned)TOK, 256, 0, stream>>>(attout, x, out_bias,
                                                     nullptr, ln1_g, ln1_b,
                                                     f1_lg, f1_lb, h1, h1ln);

  // 8) FFN1
  gemm(h1ln, f1w1b, nullptr, hidden, f1_b1, (int)TOK, FFN_, D_, D_, FFN_, FFN_,
       0, 1, 1.f, 1, 1, 0, 0, 0, 0, 0, 0);
  gemm(hidden, f1w2b, fout, nullptr, f1_b2, (int)TOK, D_, FFN_, FFN_, D_, D_,
       0, 0, 1.f, 1, 1, 0, 0, 0, 0, 0, 0);

  // 9) h2 = LN(fout + h1); h2b = bf16(h2)
  resid_ln_kernel<<<(unsigned)TOK, 256, 0, stream>>>(fout, h1, nullptr, nullptr,
                                                     ln2_g, ln2_b, nullptr,
                                                     nullptr, h2, h2b);

  // 10) conv module: pw1 (transB: pw1_w is [2D, D]) -> fp32 [token, 2D]
  gemm(h2b, pw1wb, pw1out, nullptr, pw1_b, (int)TOK, 2 * D_, D_, D_, D_,
       2 * D_, 1, 0, 1.f, 1, 1, 0, 0, 0, 0, 0, 0);

  // 11) GLU + depthwise conv + BN + Swish -> bf16
  glu_conv_kernel<<<dim3(L_, B_), 256, 0, stream>>>(pw1out, dw_w, bn_g, bn_b,
                                                    convact);

  // 12) pw2 (transB) -> fp32 cv
  gemm(convact, pw2wb, cv, nullptr, pw2_b, (int)TOK, D_, D_, D_, D_, D_, 1, 0,
       1.f, 1, 1, 0, 0, 0, 0, 0, 0);

  // 13) h3 = LN(mask*cv + h2); h3ln = LN(h3, f2_l) (bf16)
  resid_ln_kernel<<<(unsigned)TOK, 256, 0, stream>>>(cv, h2, nullptr, mask,
                                                     ln3_g, ln3_b, f2_lg,
                                                     f2_lb, h3, h3ln);

  // 14) FFN2
  gemm(h3ln, f2w1b, nullptr, hidden, f2_b1, (int)TOK, FFN_, D_, D_, FFN_, FFN_,
       0, 1, 1.f, 1, 1, 0, 0, 0, 0, 0, 0);
  gemm(hidden, f2w2b, fout, nullptr, f2_b2, (int)TOK, D_, FFN_, FFN_, D_, D_,
       0, 0, 1.f, 1, 1, 0, 0, 0, 0, 0, 0);

  // 15) h4 = LN(fout + h3) -> d_out
  resid_ln_kernel<<<(unsigned)TOK, 256, 0, stream>>>(fout, h3, nullptr, nullptr,
                                                     ln4_g, ln4_b, nullptr,
                                                     nullptr, h4_out, nullptr);
}
